// KGTBModel_42803644072106
// MI455X (gfx1250) — matrix operand order repaired and checked by reading, not compiled, not run on hardware
//
#include <hip/hip_runtime.h>

#define N_NODES 100000
#define DIM     256
#define NREL    8
#define NLAYER  3
#define NEDGE   800000
#define NPOS    32768
#define BETA_Q  0.25f

typedef __bf16 bf16_t;
typedef __bf16 v16bf __attribute__((ext_vector_type(16)));
typedef float  v8f   __attribute__((ext_vector_type(8)));

// ---------------------------------------------------------------------------
// helpers
// ---------------------------------------------------------------------------
__device__ inline float block_reduce_sum(float v, float* s) {
#pragma unroll
    for (int m = 16; m >= 1; m >>= 1) v += __shfl_xor(v, m, 32);
    int wid = threadIdx.x >> 5, lane = threadIdx.x & 31;
    if (lane == 0) s[wid] = v;
    __syncthreads();
    float r = (threadIdx.x < 8) ? s[threadIdx.x] : 0.0f;
    if (threadIdx.x < 32) {
#pragma unroll
        for (int m = 4; m >= 1; m >>= 1) r += __shfl_xor(r, m, 32);
        if (threadIdx.x == 0) s[8] = r;
    }
    __syncthreads();
    float out = s[8];
    __syncthreads();
    return out;
}

// A-fragment (16x32 bf16, wave32 layout): lane holds row lane%16,
// K contiguous {khalf*16..+15}; f32 source converted in-register.
__device__ inline v16bf load_a_frag(const float* __restrict__ row, int kk, int khalf, float scale) {
    const float* p = row + kk + (khalf << 4);
    const float4 q0 = *(const float4*)(p + 0);
    const float4 q1 = *(const float4*)(p + 4);
    const float4 q2 = *(const float4*)(p + 8);
    const float4 q3 = *(const float4*)(p + 12);
    v16bf a;
    a[0]  = (bf16_t)(q0.x * scale); a[1]  = (bf16_t)(q0.y * scale);
    a[2]  = (bf16_t)(q0.z * scale); a[3]  = (bf16_t)(q0.w * scale);
    a[4]  = (bf16_t)(q1.x * scale); a[5]  = (bf16_t)(q1.y * scale);
    a[6]  = (bf16_t)(q1.z * scale); a[7]  = (bf16_t)(q1.w * scale);
    a[8]  = (bf16_t)(q2.x * scale); a[9]  = (bf16_t)(q2.y * scale);
    a[10] = (bf16_t)(q2.z * scale); a[11] = (bf16_t)(q2.w * scale);
    a[12] = (bf16_t)(q3.x * scale); a[13] = (bf16_t)(q3.y * scale);
    a[14] = (bf16_t)(q3.z * scale); a[15] = (bf16_t)(q3.w * scale);
    return a;
}

// ---------------------------------------------------------------------------
// small utility kernels
// ---------------------------------------------------------------------------
__global__ void zero_f32(float* p, size_t n) {
    size_t i = (size_t)blockIdx.x * blockDim.x + threadIdx.x;
    if (i < n) p[i] = 0.0f;
}

__global__ void deg_kernel(const int* __restrict__ ei, float* __restrict__ deg) {
    int e = blockIdx.x * 256 + threadIdx.x;
    if (e < NEDGE) atomicAdd(&deg[ei[NEDGE + e]], 1.0f);
}

__global__ void edge_scatter(const float* __restrict__ h, const float* __restrict__ rel_emb,
                             const int* __restrict__ ei, const int* __restrict__ et,
                             float* __restrict__ agg) {
    int e = blockIdx.x, d = threadIdx.x;
    int src = ei[e], dst = ei[NEDGE + e], r = et[e];
    float m = h[(size_t)src * DIM + d] + rel_emb[(size_t)r * DIM + d];
    atomicAdd(&agg[(size_t)dst * DIM + d], m);
}

// W[d][e] (f32) -> WT[e*DIM + d] (bf16), so B-fragment K is contiguous per lane
__global__ void wt_convert(const float* __restrict__ W, bf16_t* __restrict__ WT) {
    int i = blockIdx.x * 256 + threadIdx.x;
    int d = i >> 8, e = i & 255;
    WT[(size_t)e * DIM + d] = (bf16_t)W[i];
}

// ---------------------------------------------------------------------------
// fused encoder layer: hout = relu(hin@Ws + (agg/deg)@Wa) via bf16 WMMA.
// one wave computes a 16x64 output strip (4 accumulators, A reused 4x).
// grid = (N/16)*(DIM/64)/8 blocks of 256 thr
// ---------------------------------------------------------------------------
__global__ void layer_gemm(const float* __restrict__ hin, const float* __restrict__ agg,
                           const float* __restrict__ deg,
                           const bf16_t* __restrict__ WsT, const bf16_t* __restrict__ WaT,
                           float* __restrict__ hout) {
    int wid = threadIdx.x >> 5, lane = threadIdx.x & 31;
    int tile = blockIdx.x * 8 + wid;            // 6250 * 4 = 25000 tiles
    int row0 = (tile >> 2) << 4;
    int col0 = (tile & 3) << 6;                 // 64-wide strip
    int m = lane & 15, khalf = lane >> 4;

    const float* hrow = hin + (size_t)(row0 + m) * DIM;
    const float* arow = agg + (size_t)(row0 + m) * DIM;
    float invd = 1.0f / fmaxf(deg[row0 + m], 1.0f);

    // prefetch next row-tile's A data into cache (global_prefetch_b8)
    __builtin_prefetch(hrow + (size_t)16 * DIM, 0, 0);
    __builtin_prefetch(arow + (size_t)16 * DIM, 0, 0);

    v8f acc[4] = {};
#pragma unroll
    for (int kk = 0; kk < DIM; kk += 32) {
        v16bf a1 = load_a_frag(hrow, kk, khalf, 1.0f);
        v16bf a2 = load_a_frag(arow, kk, khalf, invd);
#pragma unroll
        for (int c = 0; c < 4; c++) {
            const bf16_t* ws = WsT + (size_t)(col0 + c * 16 + m) * DIM + khalf * 16 + kk;
            const bf16_t* wa = WaT + (size_t)(col0 + c * 16 + m) * DIM + khalf * 16 + kk;
            acc[c] = __builtin_amdgcn_wmma_f32_16x16x32_bf16(false, a1, false, *(const v16bf*)ws,
                                                             (short)0, acc[c], false, false);
            acc[c] = __builtin_amdgcn_wmma_f32_16x16x32_bf16(false, a2, false, *(const v16bf*)wa,
                                                             (short)0, acc[c], false, false);
        }
    }
    int n = lane & 15;
#pragma unroll
    for (int c = 0; c < 4; c++) {
#pragma unroll
        for (int v = 0; v < 8; v++) {
            int r = row0 + v + (khalf << 3);
            hout[(size_t)r * DIM + col0 + c * 16 + n] = fmaxf(acc[c][v], 0.0f);
        }
    }
}

// ---------------------------------------------------------------------------
// row-wise layernorm (wave per row, 8 elems per lane)
// ---------------------------------------------------------------------------
__global__ void layernorm_rows(const float* __restrict__ x, const float* __restrict__ g,
                               const float* __restrict__ b, float* __restrict__ y) {
    int row = blockIdx.x * 8 + (threadIdx.x >> 5);
    int lane = threadIdx.x & 31;
    const float* xr = x + (size_t)row * DIM + lane * 8;
    float v[8];
    float s = 0.0f, s2 = 0.0f;
#pragma unroll
    for (int i = 0; i < 8; i++) { v[i] = xr[i]; s += v[i]; s2 += v[i] * v[i]; }
#pragma unroll
    for (int m = 16; m >= 1; m >>= 1) { s += __shfl_xor(s, m, 32); s2 += __shfl_xor(s2, m, 32); }
    float mu = s * (1.0f / DIM);
    float var = s2 * (1.0f / DIM) - mu * mu;
    float rstd = rsqrtf(var + 1e-5f);
    float* yr = y + (size_t)row * DIM + lane * 8;
#pragma unroll
    for (int i = 0; i < 8; i++)
        yr[i] = (v[i] - mu) * rstd * g[lane * 8 + i] + b[lane * 8 + i];
}

// ---------------------------------------------------------------------------
// residual quantizer: normalize codebook -> bf16 + |c_normalized|^2
// ---------------------------------------------------------------------------
__global__ void cb_prep(const float* __restrict__ cb, bf16_t* __restrict__ cbn,
                        float* __restrict__ nrm2) {
    int code = blockIdx.x, d = threadIdx.x;
    __shared__ float sred[9];
    float v = cb[(size_t)code * DIM + d];
    float s2 = block_reduce_sum(v * v, sred);
    float rn = 1.0f / fmaxf(sqrtf(s2), 1e-12f);
    cbn[(size_t)code * DIM + d] = (bf16_t)(v * rn);
    if (d == 0) nrm2[code] = s2 * rn * rn;
}

// argmin over codes == argmax(r·c̄ - 0.5|c̄|²): WMMA GEMM + half-wave shfl-max.
// A row (128 bf16 / lane = 64 VGPRs) is preloaded ONCE and reused for all
// K/16 code chunks -> residual traffic drops by K/16 (256x for K=4096).
__global__ void rq_argmax(const float* __restrict__ residual, const bf16_t* __restrict__ cbn,
                          const float* __restrict__ nrm2, int row_base, int nrows, int K,
                          int* __restrict__ out_idx) {
    int wid = threadIdx.x >> 5, lane = threadIdx.x & 31;
    int tile = blockIdx.x * 8 + wid;
    if (tile * 16 >= nrows) return;
    int m = lane & 15, khalf = lane >> 4, n = lane & 15;

    int rlocal = tile * 16 + m;
    if (rlocal >= nrows) rlocal = nrows - 1;
    const float* rrow = residual + (size_t)(row_base + rlocal) * DIM;

    v16bf areg[8];
#pragma unroll
    for (int k8 = 0; k8 < 8; k8++) areg[k8] = load_a_frag(rrow, k8 * 32, khalf, 1.0f);

    float best[8];
    int bidx[8];
#pragma unroll
    for (int v = 0; v < 8; v++) { best[v] = -3.4e38f; bidx[v] = 0; }

    for (int k0 = 0; k0 < K; k0 += 16) {
        v8f acc = {};
        const bf16_t* crow = cbn + (size_t)(k0 + n) * DIM + khalf * 16;
#pragma unroll
        for (int k8 = 0; k8 < 8; k8++) {
            acc = __builtin_amdgcn_wmma_f32_16x16x32_bf16(false, areg[k8], false,
                                                          *(const v16bf*)(crow + k8 * 32),
                                                          (short)0, acc, false, false);
        }
        float halfn2 = 0.5f * nrm2[k0 + n];
#pragma unroll
        for (int v = 0; v < 8; v++) {
            float s = acc[v] - halfn2;
            if (s > best[v]) { best[v] = s; bidx[v] = k0 + n; }
        }
    }
#pragma unroll
    for (int v = 0; v < 8; v++) {
        float s = best[v];
        int bi = bidx[v];
#pragma unroll
        for (int mask = 1; mask < 16; mask <<= 1) {
            float os = __shfl_xor(s, mask, 32);
            int oi = __shfl_xor(bi, mask, 32);
            if (os > s || (os == s && oi < bi)) { s = os; bi = oi; }
        }
        int rm = tile * 16 + v + (khalf << 3);
        if (n == 0 && rm < nrows) out_idx[rm] = bi;
    }
}

// apply quantization: decoded += q; residual -= q; loss += (1+β)*mean(diff²)
__global__ void rq_apply(float* __restrict__ residual, float* __restrict__ decoded,
                         const float* __restrict__ cb, const int* __restrict__ idx,
                         int row_base, int nrows, float scale, float* __restrict__ loss) {
    int row = blockIdx.x, d = threadIdx.x;
    int code = idx ? idx[row] : 0;
    float q = cb[(size_t)code * DIM + d];
    size_t off = (size_t)(row_base + row) * DIM + d;
    float r = residual[off];
    float diff = r - q;
    decoded[off] += q;
    residual[off] = diff;
    __shared__ float sred[9];
    float s = block_reduce_sum(diff * diff, sred);
    if (d == 0) atomicAdd(loss, s * scale);
}

// ---------------------------------------------------------------------------
// bilinear triple scoring + BCE: one block per triple
// ---------------------------------------------------------------------------
__global__ void triple_kernel(const float* __restrict__ decoded, const float* __restrict__ g,
                              const float* __restrict__ bln, const float* __restrict__ relw,
                              const int* __restrict__ pt, const int* __restrict__ ct,
                              float* __restrict__ loss) {
    int t = blockIdx.x, d = threadIdx.x;
    int head, rel, tail;
    float label;
    if (t < NPOS) { head = pt[t * 3]; rel = pt[t * 3 + 1]; tail = pt[t * 3 + 2]; label = 1.0f; }
    else { int u = t - NPOS; head = pt[u * 3]; rel = pt[u * 3 + 1]; tail = ct[u]; label = 0.0f; }

    __shared__ float hh[DIM], ht[DIM], sred[9];

    float xh = decoded[(size_t)head * DIM + d];
    float mu = block_reduce_sum(xh, sred) * (1.0f / DIM);
    float var = block_reduce_sum(xh * xh, sred) * (1.0f / DIM) - mu * mu;
    hh[d] = tanhf((xh - mu) * rsqrtf(var + 1e-5f) * g[d] + bln[d]);

    float xt = decoded[(size_t)tail * DIM + d];
    mu = block_reduce_sum(xt, sred) * (1.0f / DIM);
    var = block_reduce_sum(xt * xt, sred) * (1.0f / DIM) - mu * mu;
    ht[d] = tanhf((xt - mu) * rsqrtf(var + 1e-5f) * g[d] + bln[d]);
    __syncthreads();

    const float4* Wr4 = (const float4*)(relw + (size_t)rel * DIM * DIM + (size_t)d * DIM);
    float acc = 0.0f;
#pragma unroll 4
    for (int e4 = 0; e4 < DIM / 4; e4++) {
        float4 w = Wr4[e4];
        acc += w.x * ht[e4 * 4 + 0] + w.y * ht[e4 * 4 + 1]
             + w.z * ht[e4 * 4 + 2] + w.w * ht[e4 * 4 + 3];
    }
    float s = block_reduce_sum(hh[d] * acc, sred);
    if (d == 0) {
        float l = fmaxf(s, 0.0f) - s * label + log1pf(expf(-fabsf(s)));
        atomicAdd(loss, l * (1.0f / (2 * NPOS)));
    }
}

// ---------------------------------------------------------------------------
// host orchestration
// ---------------------------------------------------------------------------
extern "C" void kernel_launch(void* const* d_in, const int* in_sizes, int n_in,
                              void* d_out, int out_size, void* d_ws, size_t ws_size,
                              hipStream_t stream) {
    const float* node_emb = (const float*)d_in[0];
    const float* rel_emb  = (const float*)d_in[1];
    const float* W_self   = (const float*)d_in[2];
    const float* W_agg    = (const float*)d_in[3];
    const float* gamma    = (const float*)d_in[4];
    const float* beta     = (const float*)d_in[5];
    const float* relw     = (const float*)d_in[6];
    const float* cb[12];
    for (int i = 0; i < 12; i++) cb[i] = (const float*)d_in[7 + i];
    const int* ei = (const int*)d_in[19];
    const int* et = (const int*)d_in[20];
    const int* pt = (const int*)d_in[21];
    const int* ct = (const int*)d_in[22];
    float* out = (float*)d_out;

    const size_t ND = (size_t)N_NODES * DIM;
    char* p = (char*)d_ws;
    float* h0   = (float*)p;  p += ND * 4;
    float* h1   = (float*)p;  p += ND * 4;   // later: decoded
    float* agg  = (float*)p;  p += ND * 4;   // later: ln(h) = residual
    float* deg  = (float*)p;  p += (size_t)N_NODES * 4;
    bf16_t* wb  = (bf16_t*)p; p += (size_t)2 * DIM * DIM * 2;
    bf16_t* cbn = (bf16_t*)p; p += (size_t)4096 * DIM * 2;
    float* nrm2 = (float*)p;  p += 4096 * 4;
    int* idxbuf = (int*)p;    p += 40000 * 4;

    zero_f32<<<1, 32, 0, stream>>>(out, 1);
    zero_f32<<<(N_NODES + 255) / 256, 256, 0, stream>>>(deg, N_NODES);
    deg_kernel<<<(NEDGE + 255) / 256, 256, 0, stream>>>(ei, deg);

    // ---- encoder: 3 relational message-passing layers ----
    const float* hin = node_emb;
    float* bufs[3] = {h0, h1, h0};
    for (int l = 0; l < NLAYER; l++) {
        zero_f32<<<(int)((ND + 255) / 256), 256, 0, stream>>>(agg, ND);
        edge_scatter<<<NEDGE, DIM, 0, stream>>>(hin, rel_emb, ei, et, agg);
        wt_convert<<<DIM * DIM / 256, 256, 0, stream>>>(W_self + (size_t)l * DIM * DIM, wb);
        wt_convert<<<DIM * DIM / 256, 256, 0, stream>>>(W_agg + (size_t)l * DIM * DIM, wb + DIM * DIM);
        float* hout = bufs[l];
        layer_gemm<<<(N_NODES / 16) * (DIM / 64) / 8, 256, 0, stream>>>(hin, agg, deg, wb, wb + DIM * DIM, hout);
        hin = hout;
    }

    // ---- layernorm -> residual (agg buffer); decoded (h1) = 0 ----
    layernorm_rows<<<N_NODES / 8, 256, 0, stream>>>(hin, gamma, beta, agg);
    zero_f32<<<(int)((ND + 255) / 256), 256, 0, stream>>>(h1, ND);

    // ---- residual quantization per type: level0 K=1, then two searched levels ----
    struct TypeCfg { int base, nrows, cbi, K1, K2; };
    const TypeCfg tc[4] = {
        {    0, 40000, 0, 4096, 4096},   // poi
        {40000, 30000, 3, 4096, 4096},   // user
        {70000, 15000, 6,   64, 4096},   // region
        {85000, 15000, 9,   64, 4096},   // category
    };
    for (int t = 0; t < 4; t++) {
        int base = tc[t].base, nr = tc[t].nrows;
        float scale = (1.0f + BETA_Q) / ((float)nr * DIM);
        rq_apply<<<nr, DIM, 0, stream>>>(agg, h1, cb[tc[t].cbi], (const int*)nullptr, base, nr, scale, out);
        int Ks[2] = {tc[t].K1, tc[t].K2};
        for (int lev = 0; lev < 2; lev++) {
            const float* cbl = cb[tc[t].cbi + 1 + lev];
            int K = Ks[lev];
            cb_prep<<<K, DIM, 0, stream>>>(cbl, cbn, nrm2);
            int tiles = (nr + 15) / 16;
            rq_argmax<<<(tiles + 7) / 8, 256, 0, stream>>>(agg, cbn, nrm2, base, nr, K, idxbuf);
            rq_apply<<<nr, DIM, 0, stream>>>(agg, h1, cbl, idxbuf, base, nr, scale, out);
        }
    }

    // ---- KG bilinear scoring + BCE ----
    triple_kernel<<<2 * NPOS, DIM, 0, stream>>>(h1, gamma, beta, relw, pt, ct, out);
}
